// MultiHeadAttentionLayer_37469294691190
// MI455X (gfx1250) — matrix operand order repaired and checked by reading, not compiled, not run on hardware
//
#include <hip/hip_runtime.h>
#include <hip/hip_bf16.h>

// MHA layer for MI455X (gfx1250): all GEMMs via v_wmma_f32_16x16x32_f16
// (wave32, fp32 accumulate). Each wave owns a 16x64 output strip: one A tile
// amortized over 4 B tiles / 4 WMMAs per K-step (~26 FLOP/B from global).
// K-loops are unrolled x2 so the scheduler can overlap the next iteration's
// global_load_b128 clause with the current iteration's WMMAs.

typedef __attribute__((ext_vector_type(16))) _Float16 v16h;
typedef __attribute__((ext_vector_type(8)))  float    v8f;

#define HIDN 1024
#define NH   16
#define HD   64
#define LSEQ 2048
#define NB   2

static __device__ __forceinline__ int laneid() { return (int)(threadIdx.x & 31u); }

// ---------------------------------------------------------------------------
// A-matrix loader: 16(M) x 32(K) f16 tile, row-major, row stride ld (elems).
// A layout (contiguous K, matching B): lanes 0-15 hold row M=lane with
// K=[0..15]; lanes 16-31 hold the same row, K=[16..31].
// ---------------------------------------------------------------------------
static __device__ __forceinline__ v16h load_a_h(const _Float16* base, int ld) {
  int l  = laneid();
  int r  = l & 15;
  int kh = (l >> 4) << 4;                 // 0 or 16
  const _Float16* p = base + (size_t)r * ld + kh;
  v16h a;
  ((uint4*)&a)[0] = *(const uint4*)(p);        // K = kh .. kh+7
  ((uint4*)&a)[1] = *(const uint4*)(p + 8);    // K = kh+8 .. kh+15
  return a;
}

// Same A layout, sourced from fp32 (converted in-register to f16).
static __device__ __forceinline__ v16h load_a_f(const float* base, int ld) {
  int l  = laneid();
  int r  = l & 15;
  int kh = (l >> 4) << 4;
  const float* p = base + (size_t)r * ld + kh;
  float4 x0 = *(const float4*)(p);
  float4 x1 = *(const float4*)(p + 4);
  float4 x2 = *(const float4*)(p + 8);
  float4 x3 = *(const float4*)(p + 12);
  v16h a;
  a[0]=(_Float16)x0.x;  a[1]=(_Float16)x0.y;  a[2]=(_Float16)x0.z;  a[3]=(_Float16)x0.w;
  a[4]=(_Float16)x1.x;  a[5]=(_Float16)x1.y;  a[6]=(_Float16)x1.z;  a[7]=(_Float16)x1.w;
  a[8]=(_Float16)x2.x;  a[9]=(_Float16)x2.y;  a[10]=(_Float16)x2.z; a[11]=(_Float16)x2.w;
  a[12]=(_Float16)x3.x; a[13]=(_Float16)x3.y; a[14]=(_Float16)x3.z; a[15]=(_Float16)x3.w;
  return a;
}

// ---------------------------------------------------------------------------
// B-matrix loader: 32(K) x 16(N) f16 tile via its row-major transpose view
// (N x K, row stride ld).  B layout (contiguous K, matching A): lane n<16 holds column N=n
// with K=[0..15] in v0-7; lanes 16-31 hold the same column, K=[16..31].
// ---------------------------------------------------------------------------
static __device__ __forceinline__ v16h load_b_h(const _Float16* baseT, int ld) {
  int l  = laneid();
  int n  = l & 15;
  int kg = (l >> 4) << 4;                 // 0 or 16
  const _Float16* p = baseT + (size_t)n * ld + kg;
  v16h b;
  ((uint4*)&b)[0] = *(const uint4*)(p);
  ((uint4*)&b)[1] = *(const uint4*)(p + 8);
  return b;
}

static __device__ __forceinline__ v16h load_b_f(const float* baseT, int ld) {
  int l  = laneid();
  int n  = l & 15;
  int kg = (l >> 4) << 4;
  const float* p = baseT + (size_t)n * ld + kg;
  float4 x0 = *(const float4*)(p);
  float4 x1 = *(const float4*)(p + 4);
  float4 x2 = *(const float4*)(p + 8);
  float4 x3 = *(const float4*)(p + 12);
  v16h b;
  b[0]=(_Float16)x0.x;  b[1]=(_Float16)x0.y;  b[2]=(_Float16)x0.z;  b[3]=(_Float16)x0.w;
  b[4]=(_Float16)x1.x;  b[5]=(_Float16)x1.y;  b[6]=(_Float16)x1.z;  b[7]=(_Float16)x1.w;
  b[8]=(_Float16)x2.x;  b[9]=(_Float16)x2.y;  b[10]=(_Float16)x2.z; b[11]=(_Float16)x2.w;
  b[12]=(_Float16)x3.x; b[13]=(_Float16)x3.y; b[14]=(_Float16)x3.z; b[15]=(_Float16)x3.w;
  return b;
}

static __device__ __forceinline__ v8f wmma_f16(v16h a, v16h b, v8f c) {
  return __builtin_amdgcn_wmma_f32_16x16x32_f16(false, a, false, b, (short)0, c,
                                                false, false);
}

// ---------------------------------------------------------------------------
// Projection: y = X @ W^T + bias -> f16 [B,H,L,HD] (TR=0) or [B,H,HD,L]
// (TR=1, used for V).  One wave per 16x64 strip; TR template removes all
// epilogue branching; TR stores coalesce into one b128 per (lane, j).
// ---------------------------------------------------------------------------
template <int TR>
__global__ __launch_bounds__(256)
void proj_kernel(const float* __restrict__ X, const float* __restrict__ W,
                 const float* __restrict__ bias, _Float16* __restrict__ out) {
  int wave = (int)(blockIdx.x << 3) + (int)(threadIdx.x >> 5);
  int m0 = (wave >> 4) << 4;              // row tile (4096/16 = 256)
  int n0 = (wave & 15) << 6;              // 64-wide feature strip (1024/64 = 16)
  v8f c0 = {}, c1 = {}, c2 = {}, c3 = {};
#pragma unroll 2
  for (int k0 = 0; k0 < HIDN; k0 += 32) {
    v16h a  = load_a_f(X + (size_t)m0 * HIDN + k0, HIDN);
    v16h b0 = load_b_f(W + (size_t)(n0     ) * HIDN + k0, HIDN);
    v16h b1 = load_b_f(W + (size_t)(n0 + 16) * HIDN + k0, HIDN);
    v16h b2 = load_b_f(W + (size_t)(n0 + 32) * HIDN + k0, HIDN);
    v16h b3 = load_b_f(W + (size_t)(n0 + 48) * HIDN + k0, HIDN);
    c0 = wmma_f16(a, b0, c0);
    c1 = wmma_f16(a, b1, c1);
    c2 = wmma_f16(a, b2, c2);
    c3 = wmma_f16(a, b3, c3);
  }
  int l   = laneid();
  int nlo = l & 15;
  int mb  = m0 + ((l >> 4) << 3);
  int bi  = mb >> 11;
  int li0 = mb & (LSEQ - 1);
  v8f cj[4] = {c0, c1, c2, c3};
#pragma unroll
  for (int j = 0; j < 4; ++j) {
    int n = n0 + 16 * j + nlo;
    int h = n >> 6, d = n & 63;
    float bn = bias[n];
    if (TR) {                              // [B,H,HD,L]: 8 rows contiguous
      _Float16 tmp[8];
#pragma unroll
      for (int r = 0; r < 8; ++r) tmp[r] = (_Float16)(cj[j][r] + bn);
      size_t off = ((size_t)((bi * NH + h) * HD + d) * LSEQ + li0);
      *(uint4*)(out + off) = *(const uint4*)tmp;
    } else {                               // [B,H,L,HD]
#pragma unroll
      for (int r = 0; r < 8; ++r) {
        size_t off = ((size_t)((bi * NH + h) * LSEQ + (li0 + r)) * HD + d);
        out[off] = (_Float16)(cj[j][r] + bn);
      }
    }
  }
}

// ---------------------------------------------------------------------------
// energy[b,h,q,k] = (Q.K)/8 + alibi, replaced by -1e10 where mask==0.
// One wave per 16(q) x 64(k) strip: Q tile loaded once per d-chunk, 4 K
// tiles / 4 WMMAs; HD=64 -> 2 d-chunks, 8 WMMAs per wave.
// ---------------------------------------------------------------------------
__global__ __launch_bounds__(256)
void energy_kernel(const _Float16* __restrict__ Qh, const _Float16* __restrict__ Kh,
                   const int* __restrict__ mask, float* __restrict__ attn) {
  int wave = (int)(blockIdx.x << 3) + (int)(threadIdx.x >> 5);
  int bh = wave >> 12;                    // 0..31 (b*16+h)
  int q0 = ((wave >> 5) & 127) << 4;
  int k0 = (wave & 31) << 6;              // 64-wide k strip
  const _Float16* qb = Qh + ((size_t)bh * LSEQ + q0) * HD;
  const _Float16* kb = Kh + ((size_t)bh * LSEQ + k0) * HD;
  v8f c0 = {}, c1 = {}, c2 = {}, c3 = {};
#pragma unroll
  for (int d0 = 0; d0 < HD; d0 += 32) {
    v16h a  = load_a_h(qb + d0, HD);
    v16h b0 = load_b_h(kb + (size_t) 0 * HD + d0, HD);
    v16h b1 = load_b_h(kb + (size_t)16 * HD + d0, HD);
    v16h b2 = load_b_h(kb + (size_t)32 * HD + d0, HD);
    v16h b3 = load_b_h(kb + (size_t)48 * HD + d0, HD);
    c0 = wmma_f16(a, b0, c0);
    c1 = wmma_f16(a, b1, c1);
    c2 = wmma_f16(a, b2, c2);
    c3 = wmma_f16(a, b3, c3);
  }
  int l   = laneid();
  int nlo = l & 15;
  int qb0 = q0 + ((l >> 4) << 3);
  int b   = bh >> 4, h = bh & 15;
  float slope = 1.0f - 0.9f * (float)h * (1.0f / 15.0f);  // linspace(1.0,0.1,16)
  float* orow = attn + (size_t)bh * LSEQ * LSEQ;
  v8f cj[4] = {c0, c1, c2, c3};
#pragma unroll
  for (int j = 0; j < 4; ++j) {
    int  k  = k0 + 16 * j + nlo;
    bool mz = (mask[b * LSEQ + k] == 0);
#pragma unroll
    for (int r = 0; r < 8; ++r) {
      int q   = qb0 + r;
      float e = mz ? -1e10f
                   : (cj[j][r] * 0.125f - slope * fabsf((float)(k - q)));
      orow[(size_t)q * LSEQ + k] = e;
    }
  }
}

// ---------------------------------------------------------------------------
// In-place row softmax over Lk=2048.  One 256-thread block per row; float4
// IO, shfl_xor wave reduce + LDS cross-wave reduce.
// ---------------------------------------------------------------------------
__global__ __launch_bounds__(256)
void softmax_kernel(float* __restrict__ attn) {
  float* p = attn + (size_t)blockIdx.x * LSEQ;
  int tid  = (int)threadIdx.x;
  float4 v0 = ((const float4*)p)[tid];
  float4 v1 = ((const float4*)p)[tid + 256];
  float m = fmaxf(fmaxf(fmaxf(v0.x, v0.y), fmaxf(v0.z, v0.w)),
                  fmaxf(fmaxf(v1.x, v1.y), fmaxf(v1.z, v1.w)));
  for (int off = 16; off > 0; off >>= 1) m = fmaxf(m, __shfl_xor(m, off, 32));
  __shared__ float red[8];
  int w = tid >> 5;
  if ((tid & 31) == 0) red[w] = m;
  __syncthreads();
  m = red[0];
  for (int i = 1; i < 8; ++i) m = fmaxf(m, red[i]);
  __syncthreads();
  float4 e0, e1;
  e0.x = __expf(v0.x - m); e0.y = __expf(v0.y - m);
  e0.z = __expf(v0.z - m); e0.w = __expf(v0.w - m);
  e1.x = __expf(v1.x - m); e1.y = __expf(v1.y - m);
  e1.z = __expf(v1.z - m); e1.w = __expf(v1.w - m);
  float s = (e0.x + e0.y + e0.z + e0.w) + (e1.x + e1.y + e1.z + e1.w);
  for (int off = 16; off > 0; off >>= 1) s += __shfl_xor(s, off, 32);
  if ((tid & 31) == 0) red[w] = s;
  __syncthreads();
  s = 0.0f;
  for (int i = 0; i < 8; ++i) s += red[i];
  float inv = 1.0f / s;
  e0.x *= inv; e0.y *= inv; e0.z *= inv; e0.w *= inv;
  e1.x *= inv; e1.y *= inv; e1.z *= inv; e1.w *= inv;
  ((float4*)p)[tid]       = e0;
  ((float4*)p)[tid + 256] = e1;
}

// ---------------------------------------------------------------------------
// xattn[b,h,q,d] = attn @ V.  One wave handles a full 16(q) x 64(d=HD) strip:
// A (probs, fp32->f16 inline) loaded once per k-step, 4 V tiles / 4 WMMAs.
// ---------------------------------------------------------------------------
__global__ __launch_bounds__(256)
void attnv_kernel(const float* __restrict__ attn, const _Float16* __restrict__ Vt,
                  _Float16* __restrict__ xh) {
  int wave = (int)(blockIdx.x << 3) + (int)(threadIdx.x >> 5);
  int bh = wave >> 7;                     // 0..31
  int q0 = (wave & 127) << 4;
  const float*    ab = attn + ((size_t)bh * LSEQ + q0) * LSEQ;
  const _Float16* vb = Vt   + (size_t)bh * HD * LSEQ;
  v8f c0 = {}, c1 = {}, c2 = {}, c3 = {};
#pragma unroll 2
  for (int k0 = 0; k0 < LSEQ; k0 += 32) {
    v16h a  = load_a_f(ab + k0, LSEQ);
    v16h b0 = load_b_h(vb + (size_t) 0 * LSEQ + k0, LSEQ);
    v16h b1 = load_b_h(vb + (size_t)16 * LSEQ + k0, LSEQ);
    v16h b2 = load_b_h(vb + (size_t)32 * LSEQ + k0, LSEQ);
    v16h b3 = load_b_h(vb + (size_t)48 * LSEQ + k0, LSEQ);
    c0 = wmma_f16(a, b0, c0);
    c1 = wmma_f16(a, b1, c1);
    c2 = wmma_f16(a, b2, c2);
    c3 = wmma_f16(a, b3, c3);
  }
  int l   = laneid();
  int nlo = l & 15;
  int qb0 = q0 + ((l >> 4) << 3);
  v8f cj[4] = {c0, c1, c2, c3};
#pragma unroll
  for (int j = 0; j < 4; ++j) {
    int d = 16 * j + nlo;
#pragma unroll
    for (int r = 0; r < 8; ++r)
      xh[((size_t)bh * LSEQ + (qb0 + r)) * HD + d] = (_Float16)cj[j][r];
  }
}

// ---------------------------------------------------------------------------
// out = xattn(perm to [B,L,H*HD]) @ Wo^T + bo, fp32.  [B,H,L,HD] re-indexed
// per 32-wide K chunk (each chunk stays inside one head row).
// ---------------------------------------------------------------------------
__global__ __launch_bounds__(256)
void outproj_kernel(const _Float16* __restrict__ xh, const float* __restrict__ Wo,
                    const float* __restrict__ bo, float* __restrict__ out) {
  int wave = (int)(blockIdx.x << 3) + (int)(threadIdx.x >> 5);
  int m0 = (wave >> 4) << 4;
  int n0 = (wave & 15) << 6;
  int b  = m0 >> 11, l0 = m0 & (LSEQ - 1);
  v8f c0 = {}, c1 = {}, c2 = {}, c3 = {};
#pragma unroll 2
  for (int cc = 0; cc < HIDN; cc += 32) {
    int h = cc >> 6, dofs = cc & 63;
    const _Float16* ab = xh + ((size_t)(b * NH + h) * LSEQ + l0) * HD + dofs;
    v16h a  = load_a_h(ab, HD);
    v16h b0 = load_b_f(Wo + (size_t)(n0     ) * HIDN + cc, HIDN);
    v16h b1 = load_b_f(Wo + (size_t)(n0 + 16) * HIDN + cc, HIDN);
    v16h b2 = load_b_f(Wo + (size_t)(n0 + 32) * HIDN + cc, HIDN);
    v16h b3 = load_b_f(Wo + (size_t)(n0 + 48) * HIDN + cc, HIDN);
    c0 = wmma_f16(a, b0, c0);
    c1 = wmma_f16(a, b1, c1);
    c2 = wmma_f16(a, b2, c2);
    c3 = wmma_f16(a, b3, c3);
  }
  int l   = laneid();
  int nlo = l & 15;
  int mb  = m0 + ((l >> 4) << 3);
  v8f cj[4] = {c0, c1, c2, c3};
#pragma unroll
  for (int j = 0; j < 4; ++j) {
    int n = n0 + 16 * j + nlo;
    float bn = bo[n];
#pragma unroll
    for (int r = 0; r < 8; ++r)
      out[(size_t)(mb + r) * HIDN + n] = cj[j][r] + bn;
  }
}

// ---------------------------------------------------------------------------
extern "C" void kernel_launch(void* const* d_in, const int* in_sizes, int n_in,
                              void* d_out, int out_size, void* d_ws, size_t ws_size,
                              hipStream_t stream) {
  const float* query = (const float*)d_in[0];
  const float* key   = (const float*)d_in[1];
  const float* value = (const float*)d_in[2];
  const int*   mask  = (const int*)d_in[3];
  const float* Wq = (const float*)d_in[4];
  const float* bq = (const float*)d_in[5];
  const float* Wk = (const float*)d_in[6];
  const float* bk = (const float*)d_in[7];
  const float* Wv = (const float*)d_in[8];
  const float* bv = (const float*)d_in[9];
  const float* Wo = (const float*)d_in[10];
  const float* bo = (const float*)d_in[11];

  float* xout = (float*)d_out;                         // [B,L,HID]
  float* attn = xout + (size_t)NB * LSEQ * HIDN;       // [B,H,Lq,Lk]

  const size_t qkv = (size_t)NB * NH * LSEQ * HD;      // 4M halves each
  _Float16* Qh = (_Float16*)d_ws;                      // [B,H,L,HD]
  _Float16* Kh = Qh + qkv;                             // [B,H,L,HD]
  _Float16* Vt = Kh + qkv;                             // [B,H,HD,L]
  _Float16* Xh = Vt + qkv;                             // [B,H,L,HD]

  dim3 blk(256);
  proj_kernel<0><<<512,   blk, 0, stream>>>(query, Wq, bq, Qh);
  proj_kernel<0><<<512,   blk, 0, stream>>>(key,   Wk, bk, Kh);
  proj_kernel<1><<<512,   blk, 0, stream>>>(value, Wv, bv, Vt);
  energy_kernel<<<16384,  blk, 0, stream>>>(Qh, Kh, mask, attn);
  softmax_kernel<<<65536, blk, 0, stream>>>(attn);
  attnv_kernel<<<512,     blk, 0, stream>>>(attn, Vt, Xh);
  outproj_kernel<<<512,   blk, 0, stream>>>(Xh, Wo, bo, xout);
}
